// recurrent_noisy_42786464202992
// MI455X (gfx1250) — matrix operand order repaired and checked by reading, not compiled, not run on hardware
//
#include <hip/hip_runtime.h>
#include <hip/hip_bf16.h>

// Problem dims (match reference)
#define B_ 128
#define T_ 512
#define I_ 256
#define H_ 1024
#define O_ 64

typedef __attribute__((ext_vector_type(16))) _Float16 v16h;
typedef __attribute__((ext_vector_type(8)))  _Float16 v8h;
typedef __attribute__((ext_vector_type(8)))  float    v8f;
typedef __attribute__((ext_vector_type(4)))  float    v4f;

static __device__ __forceinline__ v16h cat16(v8h lo, v8h hi) {
  v16h a;
#pragma unroll
  for (int i = 0; i < 8; ++i) { a[i] = lo[i]; a[i + 8] = hi[i]; }
  return a;
}

// A fragment (16x32, f16): lanes 0-15 need K {0..7,16..23},
// lanes 16-31 need K {8..15,24..31} of row (lane&15). Two b128 loads.
static __device__ __forceinline__ v16h load_a_f16(const _Float16* __restrict__ rowptr,
                                                  int k0, int lane) {
  int kb = k0 + ((lane >> 4) << 3);
  v8h lo = *(const v8h*)(rowptr + kb);
  v8h hi = *(const v8h*)(rowptr + kb + 16);
  return cat16(lo, hi);
}

// A fragment from f32 source with inline f32->f16 convert.
static __device__ __forceinline__ v16h load_a_cvt(const float* __restrict__ rowptr,
                                                  int k0, int lane) {
  int kb = k0 + ((lane >> 4) << 3);
  v4f f0 = *(const v4f*)(rowptr + kb);
  v4f f1 = *(const v4f*)(rowptr + kb + 4);
  v4f f2 = *(const v4f*)(rowptr + kb + 16);
  v4f f3 = *(const v4f*)(rowptr + kb + 20);
  v16h a;
#pragma unroll
  for (int i = 0; i < 4; ++i) {
    a[i]      = (_Float16)f0[i];
    a[i + 4]  = (_Float16)f1[i];
    a[i + 8]  = (_Float16)f2[i];
    a[i + 12] = (_Float16)f3[i];
  }
  return a;
}

// B fragment from pre-packed buffer: fragment = 32 lanes x 16 f16, contiguous.
static __device__ __forceinline__ v16h load_b_frag(const _Float16* __restrict__ packed,
                                                   int frag, int lane) {
  const _Float16* p = packed + (((size_t)frag << 5) + (size_t)lane) * 16;
  v8h lo = *(const v8h*)p;
  v8h hi = *(const v8h*)(p + 8);
  return cat16(lo, hi);
}

// ---------------- prep kernels ----------------

// generic f32 -> f16 elementwise (x and hidden0)
__global__ void cvt_f16_kernel(const float* __restrict__ src, _Float16* __restrict__ dst, int n) {
  int id = blockIdx.x * blockDim.x + threadIdx.x;
  if (id < n) dst[id] = (_Float16)src[id];
}

// Pack W (row-major [N][K], used as B[k][n] = W[n][k]) into WMMA B-fragment order.
__global__ void pack_b_kernel(const float* __restrict__ W, _Float16* __restrict__ dst,
                              int N, int K) {
  int id = blockIdx.x * blockDim.x + threadIdx.x;
  if (id >= N * K) return;
  int e    = id & 15;
  int lane = (id >> 4) & 31;
  int frag = id >> 9;
  int KT = K >> 5;
  int kt = frag % KT;
  int jt = frag / KT;
  int n = jt * 16 + (lane & 15);
  int k = kt * 32 + ((lane >> 4) << 3) + (e & 7) + ((e >> 3) << 4);
  dst[id] = (_Float16)W[(size_t)n * K + k];
}

// ---------------- drive GEMM: net[:,t,:] = x @ W_in^T + b_in + sigma*noise ----------------
// M = B*T = 65536 rows, N = H, K = I. One 16x16 tile per wave, 2 accumulation chains.
__global__ void drive_kernel(const _Float16* __restrict__ xh,
                             const _Float16* __restrict__ winP,
                             const float* __restrict__ b_in,
                             const float* __restrict__ noise,
                             const float* __restrict__ sigma_p,
                             float* __restrict__ net) {
  const int lane = threadIdx.x & 31;
  const int wid  = threadIdx.x >> 5;
  const int tid  = blockIdx.x * 4 + wid;
  const int jt = tid & 63;                    // H/16 = 64 col tiles
  const int mt = tid >> 6;                    // 4096 row tiles
  const int KT = I_ / 32;                     // 8

  const _Float16* arow = xh + (size_t)(mt * 16 + (lane & 15)) * I_;
  v8f c0 = {}, c1 = {};
#pragma unroll
  for (int kt = 0; kt < KT; kt += 2) {
    v16h a0 = load_a_f16(arow, kt * 32, lane);
    v16h b0 = load_b_frag(winP, jt * KT + kt, lane);
    v16h a1 = load_a_f16(arow, (kt + 1) * 32, lane);
    v16h b1 = load_b_frag(winP, jt * KT + kt + 1, lane);
    c0 = __builtin_amdgcn_wmma_f32_16x16x32_f16(false, a0, false, b0, (short)0, c0, false, false);
    c1 = __builtin_amdgcn_wmma_f32_16x16x32_f16(false, a1, false, b1, (short)0, c1, false, false);
  }
  const float sigma = sigma_p[0];
  const int col   = jt * 16 + (lane & 15);
  const int rbase = mt * 16 + ((lane >> 4) << 3);
  const float bb = b_in[col];
#pragma unroll
  for (int r = 0; r < 8; ++r) {
    size_t idx = (size_t)(rbase + r) * H_ + col;   // [B*T][H] flat == [B][T][H]
    net[idx] = c0[r] + c1[r] + bb + sigma * noise[idx];
  }
}

// ---------------- recurrent step (in place on net_units) ----------------
// h_new[b,j] = relu(net[b,t,j] + sum_k h_prev[b,k]*W_h[j,k] + b_h[j])
// hprev: f16 [B,H] shadow of previous hidden state; hnext: f16 [B,H] written here.
// 4 independent WMMA accumulation chains to fill the XDL pipe at ~1 wave/SIMD.
__global__ void scan_step_kernel(const _Float16* __restrict__ hprev,
                                 const _Float16* __restrict__ whP,
                                 const float* __restrict__ b_h,
                                 float* __restrict__ net,
                                 _Float16* __restrict__ hnext,
                                 int t) {
  const int lane = threadIdx.x & 31;
  const int wid  = threadIdx.x >> 5;
  const int tid  = blockIdx.x * 4 + wid;     // 0..511
  const int jt = tid & 63;
  const int mt = tid >> 6;                   // 0..7 (B/16)
  const int KT = H_ / 32;                    // 32

  const _Float16* arow = hprev + (size_t)(mt * 16 + (lane & 15)) * H_;

  const int col   = jt * 16 + (lane & 15);
  const int rbase = mt * 16 + ((lane >> 4) << 3);
  const float bb = b_h[col];

  v8f c0, c1 = {}, c2 = {}, c3 = {};
#pragma unroll
  for (int r = 0; r < 8; ++r) {
    size_t idx = ((size_t)(rbase + r) * T_ + t) * H_ + col;
    c0[r] = net[idx] + bb;                   // C-init = drive + bias
  }
#pragma unroll 2
  for (int kt = 0; kt < KT; kt += 4) {
    v16h a0 = load_a_f16(arow, (kt + 0) * 32, lane);
    v16h b0 = load_b_frag(whP, jt * KT + kt + 0, lane);
    v16h a1 = load_a_f16(arow, (kt + 1) * 32, lane);
    v16h b1 = load_b_frag(whP, jt * KT + kt + 1, lane);
    v16h a2 = load_a_f16(arow, (kt + 2) * 32, lane);
    v16h b2 = load_b_frag(whP, jt * KT + kt + 2, lane);
    v16h a3 = load_a_f16(arow, (kt + 3) * 32, lane);
    v16h b3 = load_b_frag(whP, jt * KT + kt + 3, lane);
    c0 = __builtin_amdgcn_wmma_f32_16x16x32_f16(false, a0, false, b0, (short)0, c0, false, false);
    c1 = __builtin_amdgcn_wmma_f32_16x16x32_f16(false, a1, false, b1, (short)0, c1, false, false);
    c2 = __builtin_amdgcn_wmma_f32_16x16x32_f16(false, a2, false, b2, (short)0, c2, false, false);
    c3 = __builtin_amdgcn_wmma_f32_16x16x32_f16(false, a3, false, b3, (short)0, c3, false, false);
  }
#pragma unroll
  for (int r = 0; r < 8; ++r) {
    size_t idx = ((size_t)(rbase + r) * T_ + t) * H_ + col;
    float v = c0[r] + c1[r] + c2[r] + c3[r];
    v = v > 0.0f ? v : 0.0f;                 // relu
    net[idx] = v;                            // f32 net_units (in place)
    hnext[(size_t)(rbase + r) * H_ + col] = (_Float16)v;  // f16 shadow for next step
  }
}

// ---------------- readout GEMM: rou = net @ W_fc^T + b_fc ----------------
// M = B*T, N = O = 64 (4 col tiles per wave, A reuse), K = H.
__global__ void readout_kernel(const float* __restrict__ net,
                               const _Float16* __restrict__ wfcP,
                               const float* __restrict__ b_fc,
                               float* __restrict__ rou) {
  const int lane = threadIdx.x & 31;
  const int wid  = threadIdx.x >> 5;
  const int mt   = blockIdx.x * 4 + wid;     // 0..4095
  const int KT = H_ / 32;                    // 32

  const float* arow = net + (size_t)(mt * 16 + (lane & 15)) * H_;
  v8f c0 = {}, c1 = {}, c2 = {}, c3 = {};
#pragma unroll 4
  for (int kt = 0; kt < KT; ++kt) {
    v16h a = load_a_cvt(arow, kt * 32, lane);
    v16h b0 = load_b_frag(wfcP, 0 * KT + kt, lane);
    v16h b1 = load_b_frag(wfcP, 1 * KT + kt, lane);
    v16h b2 = load_b_frag(wfcP, 2 * KT + kt, lane);
    v16h b3 = load_b_frag(wfcP, 3 * KT + kt, lane);
    c0 = __builtin_amdgcn_wmma_f32_16x16x32_f16(false, a, false, b0, (short)0, c0, false, false);
    c1 = __builtin_amdgcn_wmma_f32_16x16x32_f16(false, a, false, b1, (short)0, c1, false, false);
    c2 = __builtin_amdgcn_wmma_f32_16x16x32_f16(false, a, false, b2, (short)0, c2, false, false);
    c3 = __builtin_amdgcn_wmma_f32_16x16x32_f16(false, a, false, b3, (short)0, c3, false, false);
  }
  const int rbase = mt * 16 + ((lane >> 4) << 3);
#pragma unroll
  for (int jt = 0; jt < 4; ++jt) {
    v8f c = (jt == 0) ? c0 : (jt == 1) ? c1 : (jt == 2) ? c2 : c3;
    int col = jt * 16 + (lane & 15);
    float bb = b_fc[col];
#pragma unroll
    for (int r = 0; r < 8; ++r) {
      rou[(size_t)(rbase + r) * O_ + col] = c[r] + bb;
    }
  }
}

// ---------------- finalize: out = rou[:, t_dec]; hidden_final = net[:, T-1] ----------------
__global__ void finalize_kernel(const float* __restrict__ net,
                                const float* __restrict__ rou,
                                const int* __restrict__ t_dec_p,
                                float* __restrict__ out,
                                float* __restrict__ hidden_final) {
  int id = blockIdx.x * blockDim.x + threadIdx.x;
  int td = t_dec_p[0];
  if (id < B_ * O_) {
    int b = id / O_, o = id % O_;
    out[id] = rou[((size_t)b * T_ + td) * O_ + o];
  }
  int id2 = id - B_ * O_;
  if (id2 >= 0 && id2 < B_ * H_) {
    int b = id2 / H_, h = id2 % H_;
    hidden_final[id2] = net[((size_t)b * T_ + (T_ - 1)) * H_ + h];
  }
}

// ---------------- host side ----------------

extern "C" void kernel_launch(void* const* d_in, const int* in_sizes, int n_in,
                              void* d_out, int out_size, void* d_ws, size_t ws_size,
                              hipStream_t stream) {
  (void)in_sizes; (void)n_in; (void)out_size; (void)ws_size;

  const float* x        = (const float*)d_in[0];   // [B,T,I]
  const float* noise    = (const float*)d_in[1];   // [B,T,H]
  const float* hidden0  = (const float*)d_in[2];   // [B,H]
  const float* sigma    = (const float*)d_in[3];   // scalar
  const float* W_in     = (const float*)d_in[4];   // [H,I]
  const float* b_in     = (const float*)d_in[5];   // [H]
  const float* W_h      = (const float*)d_in[6];   // [H,H]
  const float* b_h      = (const float*)d_in[7];   // [H]
  const float* W_fc     = (const float*)d_in[8];   // [O,H]
  const float* b_fc     = (const float*)d_in[9];   // [O]
  const int*   t_dec    = (const int*)d_in[10];

  float* out_f = (float*)d_out;
  float* out_v        = out_f;                               // [B,O]
  float* hidden_final = out_f + (size_t)B_ * O_;             // [B,H]
  float* net          = hidden_final + (size_t)B_ * H_;      // [B,T,H]
  float* rou          = net + (size_t)B_ * T_ * H_;          // [B,T,O]

  // Workspace layout (~36.8 MB total)
  char* w = (char*)d_ws;
  _Float16* xh    = (_Float16*)w;                                   // 33,554,432 B
  _Float16* winP  = (_Float16*)(w + 33554432);                      //    524,288 B
  _Float16* whP   = (_Float16*)(w + 34078720);                      //  2,097,152 B
  _Float16* wfcP  = (_Float16*)(w + 36175872);                      //    131,072 B
  _Float16* hbuf0 = (_Float16*)(w + 36306944);                      //    262,144 B
  _Float16* hbuf1 = (_Float16*)(w + 36569088);                      //    262,144 B

  // 1) precision conversion + weight fragment packing
  {
    int n = B_ * T_ * I_;
    cvt_f16_kernel<<<(n + 255) / 256, 256, 0, stream>>>(x, xh, n);
    cvt_f16_kernel<<<(B_ * H_ + 255) / 256, 256, 0, stream>>>(hidden0, hbuf0, B_ * H_);
    pack_b_kernel<<<(H_ * I_ + 255) / 256, 256, 0, stream>>>(W_in, winP, H_, I_);
    pack_b_kernel<<<(H_ * H_ + 255) / 256, 256, 0, stream>>>(W_h, whP, H_, H_);
    pack_b_kernel<<<(O_ * H_ + 255) / 256, 256, 0, stream>>>(W_fc, wfcP, O_, H_);
  }

  // 2) drive GEMM directly into net_units region (in-place scratch elimination)
  {
    int tiles = (B_ * T_ / 16) * (H_ / 16);   // 4096 * 64
    drive_kernel<<<tiles / 4, 128, 0, stream>>>(xh, winP, b_in, noise, sigma, net);
  }

  // 3) 512 sequential recurrent steps, in place on net_units, f16 h double-buffered
  {
    int tiles = (B_ / 16) * (H_ / 16);        // 512 tiles -> 128 blocks of 4 waves
    for (int t = 0; t < T_; ++t) {
      _Float16* hp = (t & 1) ? hbuf1 : hbuf0;
      _Float16* hn = (t & 1) ? hbuf0 : hbuf1;
      scan_step_kernel<<<tiles / 4, 128, 0, stream>>>(hp, whP, b_h, net, hn, t);
    }
  }

  // 4) readout GEMM
  {
    int mtiles = B_ * T_ / 16;                // 4096
    readout_kernel<<<mtiles / 4, 128, 0, stream>>>(net, wfcP, b_fc, rou);
  }

  // 5) out + hidden_final extraction
  {
    int n = B_ * O_ + B_ * H_;
    finalize_kernel<<<(n + 255) / 256, 256, 0, stream>>>(net, rou, t_dec, out_v, hidden_final);
  }
}
